// MultiHeadSelfAttention_42417097017157
// MI455X (gfx1250) — matrix operand order repaired and checked by reading, not compiled, not run on hardware
//
#include <hip/hip_runtime.h>

// ---------------------------------------------------------------------------
// MHA for MI455X (gfx1250, wave32, WMMA). f16 operands, f32 accumulate.
// Attention uses transposed-S formulation: lane L%16 owns one query column,
// so online-softmax state (m, l, alpha) is lane-private scalar state and the
// per-chunk reductions need only one shfl_xor(16) each. K/V tiles are staged
// once per block in LDS via async DMA (shared by all 4 waves).
// ---------------------------------------------------------------------------

typedef __attribute__((ext_vector_type(16))) _Float16 v16h;
typedef __attribute__((ext_vector_type(8)))  _Float16 v8h;
typedef __attribute__((ext_vector_type(8)))  float    v8f;

#define DEV __device__ __forceinline__

constexpr int Cdim = 1024;  // embedding dim = K = N of projections
constexpr int Tdim = 2048;  // sequence length
constexpr int Hn   = 16;    // heads
constexpr int Dh   = 64;    // head dim

// Async global->LDS DMA (gfx1250) via inline asm (ISA §15.18, opcode 98).
#if defined(__gfx1250__)
#define HAVE_ASYNC_LDS 1
#endif

#ifdef HAVE_ASYNC_LDS
DEV void async_b128(const void* gptr, const void* lptr) {
  // Generic LDS pointer: addr[31:0] is the LDS offset (flat-aperture rule).
  const unsigned ldsaddr = (unsigned)(uintptr_t)lptr;
  asm volatile("global_load_async_to_lds_b128 %0, %1, off"
               :
               : "v"(ldsaddr), "v"(gptr)
               : "memory");
}
DEV void wait_asynccnt0() {
  asm volatile("s_wait_asynccnt 0" ::: "memory");
}
#endif

DEV v16h ldfrag(const _Float16* p0, const _Float16* p1) {
  union { v16h v; v8h h[2]; } u;
  u.h[0] = *(const v8h*)p0;
  u.h[1] = *(const v8h*)p1;
  return u.v;
}

DEV v8f wmma_f16(v16h a, v16h b, v8f c) {
  // D = A(16x32 f16) * B(32x16 f16) + C(16x16 f32)
  return __builtin_amdgcn_wmma_f32_16x16x32_f16(false, a, false, b, (short)0, c,
                                                false, false);
}

// ---------------------------------------------------------------------------
// f32 -> f16 conversion (with scale folding)
// ---------------------------------------------------------------------------
__global__ void k_convert_f16(const float* __restrict__ in,
                              _Float16* __restrict__ out, float scale, int n) {
  int i = blockIdx.x * blockDim.x + threadIdx.x;
  if (i < n) out[i] = (_Float16)(in[i] * scale);
}

// ---------------------------------------------------------------------------
// Tiled WMMA GEMM:  y[m,n] = sum_k A16[m,k] * W16[n,k] + biasScale*bias[n]
// Block = 128 threads (4 waves), 64x64 output tile, K-step 32, double-buffered
// LDS staging (async DMA when available).
// MODE 0: store f16 row-major [M, C]
// MODE 1: store f16 transposed per head: Vt[b, h, d, t]  (packed b128 stores)
// MODE 2: store f32 row-major [M, C]
// ---------------------------------------------------------------------------
template <int MODE>
__global__ __launch_bounds__(128) void k_gemm(const _Float16* __restrict__ A,
                                              const _Float16* __restrict__ W,
                                              const float* __restrict__ bias,
                                              float biasScale,
                                              void* __restrict__ outp) {
  constexpr int LDT = 40;  // padded row stride (halves) -> conflict-free b128
  __shared__ alignas(16) _Float16 sA[2][64][LDT];
  __shared__ alignas(16) _Float16 sW[2][64][LDT];

  const int tid = threadIdx.x;
  const int w = tid >> 5, L = tid & 31, g = L >> 4, r16 = L & 15;
  const int mBase = blockIdx.x * 64;
  const int nBase = blockIdx.y * 64;

  // Cooperative staging: 128 threads x 2 chunks of 16B each per tile.
  const int row0 = tid >> 2, sub0 = tid & 3;          // chunks 0..127
  const int row1 = (tid + 128) >> 2, sub1 = tid & 3;  // chunks 128..255

  v8f acc[4] = {};
  constexpr int NSTEP = Cdim / 32;

#ifdef HAVE_ASYNC_LDS
  auto async_cp = [&](int buf, int kk) {
    async_b128(&A[(size_t)(mBase + row0) * Cdim + kk + sub0 * 8],
               &sA[buf][row0][sub0 * 8]);
    async_b128(&A[(size_t)(mBase + row1) * Cdim + kk + sub1 * 8],
               &sA[buf][row1][sub1 * 8]);
    async_b128(&W[(size_t)(nBase + row0) * Cdim + kk + sub0 * 8],
               &sW[buf][row0][sub0 * 8]);
    async_b128(&W[(size_t)(nBase + row1) * Cdim + kk + sub1 * 8],
               &sW[buf][row1][sub1 * 8]);
  };
  async_cp(0, 0);
#else
  v8h ra0, ra1, rw0, rw1;
  auto gload = [&](int kk) {
    ra0 = *(const v8h*)&A[(size_t)(mBase + row0) * Cdim + kk + sub0 * 8];
    ra1 = *(const v8h*)&A[(size_t)(mBase + row1) * Cdim + kk + sub1 * 8];
    rw0 = *(const v8h*)&W[(size_t)(nBase + row0) * Cdim + kk + sub0 * 8];
    rw1 = *(const v8h*)&W[(size_t)(nBase + row1) * Cdim + kk + sub1 * 8];
  };
  auto stage = [&](int buf) {
    *(v8h*)&sA[buf][row0][sub0 * 8] = ra0;
    *(v8h*)&sA[buf][row1][sub1 * 8] = ra1;
    *(v8h*)&sW[buf][row0][sub0 * 8] = rw0;
    *(v8h*)&sW[buf][row1][sub1 * 8] = rw1;
  };
  gload(0);
  stage(0);
#endif

  for (int s = 0; s < NSTEP; ++s) {
    const int cur = s & 1;
    const bool more = (s + 1) < NSTEP;
#ifdef HAVE_ASYNC_LDS
    wait_asynccnt0();  // this wave's DMA landed in LDS
    __syncthreads();   // everyone's DMA landed
    if (more) async_cp(1 - cur, (s + 1) * 32);  // overlap next tile with math
#else
    __syncthreads();
    if (more) gload((s + 1) * 32);
    if (s + 2 < NSTEP) {
      __builtin_prefetch(&A[(size_t)(mBase + row0) * Cdim + (s + 2) * 32], 0, 0);
      __builtin_prefetch(&W[(size_t)(nBase + row0) * Cdim + (s + 2) * 32], 0, 0);
    }
#endif
    // A fragment: wave w rows 16w..16w+15; lane holds row r16,
    // K = 16g+0..15 contiguous.
    const _Float16* arow = &sA[cur][16 * w + r16][0];
    v16h af = ldfrag(arow + 16 * g, arow + 16 * g + 8);
#pragma unroll
    for (int j = 0; j < 4; ++j) {
      // B fragment: lane holds column n = j*16+r16 of B = row n of W tile,
      // K = 16g+0..15 contiguous.
      const _Float16* brow = &sW[cur][16 * j + r16][16 * g];
      v16h bf = ldfrag(brow, brow + 8);
      acc[j] = wmma_f16(af, bf, acc[j]);
    }
#ifndef HAVE_ASYNC_LDS
    if (more) stage(1 - cur);
#endif
  }

  // Epilogue: bias + store. C/D layout: VGPR r, lane-half g -> row 8g+r,
  // lane%16 -> column.
#pragma unroll
  for (int j = 0; j < 4; ++j) {
    const int ncol = nBase + j * 16 + r16;
    const float bval = biasScale * bias[ncol];
    if (MODE == 1) {
      // Vt[b, h, d, t]: fixed (h,d)=ncol, t contiguous over r -> one b128.
      const int b = mBase >> 11;                       // tile never crosses b
      const int t0 = (mBase & (Tdim - 1)) + 16 * w + 8 * g;
      const int hh = ncol >> 6, dd = ncol & (Dh - 1);
      v8h vh;
#pragma unroll
      for (int r = 0; r < 8; ++r) vh[r] = (_Float16)(acc[j][r] + bval);
      *(v8h*)&((_Float16*)outp)[(((size_t)(b * Hn + hh)) * Dh + dd) * Tdim +
                                t0] = vh;
    } else {
#pragma unroll
      for (int r = 0; r < 8; ++r) {
        const int mrow = mBase + 16 * w + 8 * g + r;
        const float val = acc[j][r] + bval;
        if (MODE == 0) {
          ((_Float16*)outp)[(size_t)mrow * Cdim + ncol] = (_Float16)val;
        } else {
          ((float*)outp)[(size_t)mrow * Cdim + ncol] = val;
        }
      }
    }
  }
}

// ---------------------------------------------------------------------------
// Flash-attention, transposed formulation, LDS-staged K/V.
// Block = (64-query tile, h, b), 4 waves x 16 query rows.
//   S^T = K (16keys x 32d, A-op from LDS) @ Q^T (32d x 16q, B-op)
//   O^T = V^T (16d x 32keys, A-op from LDS) @ P^T (32keys x 16q, B-op, LDS)
// K/V 64-key chunks are DMA'd to LDS once per block (4x traffic reduction vs
// per-wave global loads), double-buffered; one barrier per chunk.
// ---------------------------------------------------------------------------
__global__ __launch_bounds__(128) void k_attention(
    const _Float16* __restrict__ Q, const _Float16* __restrict__ Kf,
    const _Float16* __restrict__ Vt, _Float16* __restrict__ O16) {
  constexpr int LDP = 72;  // padded row stride (halves)
  __shared__ alignas(16) _Float16 sK[2][64][LDP];  // [key][d]
  __shared__ alignas(16) _Float16 sV[2][64][LDP];  // [d][key]
  __shared__ alignas(16) _Float16 sP[4][16][LDP];  // per-wave P[query][key]

  const int tid = threadIdx.x;
  const int w = tid >> 5, L = tid & 31, g = L >> 4, r16 = L & 15;
  const int qBase = blockIdx.x * 64 + w * 16;
  const int h = blockIdx.y;
  const int bi = blockIdx.z;
  const size_t kfbase = (size_t)(bi * Tdim) * Cdim + h * Dh;   // + key*Cdim
  const size_t vtbase = (size_t)(bi * Hn + h) * Dh * Tdim;     // + d*Tdim

  // B-operand fragments of Q^T: lane's column = query r16 = one Q row.
  v16h bq[2];
#pragma unroll
  for (int dc = 0; dc < 2; ++dc) {
    const _Float16* qrow =
        &Q[(size_t)(bi * Tdim + qBase + r16) * Cdim + h * Dh + 32 * dc + 8 * g];
    bq[dc] = ldfrag(qrow, qrow + 16);
  }

  float mI = -1.0e30f, lI = 0.0f;
  v8f ot[4] = {};  // O^T accum: row d = 16n + 8g + r, col = query r16

  constexpr int NC = Tdim / 64;  // 32 key chunks

#ifdef HAVE_ASYNC_LDS
  // 512 16B-chunks per tile (64 rows x 8 subs); 128 threads x 4 each.
  auto stage_async = [&](int buf, int kBase) {
#pragma unroll
    for (int i = 0; i < 4; ++i) {
      const int c = tid + 128 * i;
      const int row = c >> 3, sub = c & 7;
      async_b128(&Kf[kfbase + (size_t)(kBase + row) * Cdim + sub * 8],
                 &sK[buf][row][sub * 8]);
      async_b128(&Vt[vtbase + (size_t)row * Tdim + kBase + sub * 8],
                 &sV[buf][row][sub * 8]);
    }
  };
  stage_async(0, 0);
#endif

  for (int kc = 0; kc < NC; ++kc) {
    const int kBase = kc * 64;
#ifdef HAVE_ASYNC_LDS
    const int cur = kc & 1;
    wait_asynccnt0();  // this wave's DMA landed
    __syncthreads();   // everyone's DMA landed
    if (kc + 1 < NC) stage_async(1 - cur, kBase + 64);  // overlap with math
#else
    const int cur = 0;
    v8h rk[4], rv[4];
#pragma unroll
    for (int i = 0; i < 4; ++i) {
      const int c = tid + 128 * i;
      const int row = c >> 3, sub = c & 7;
      rk[i] = *(const v8h*)&Kf[kfbase + (size_t)(kBase + row) * Cdim + sub * 8];
      rv[i] = *(const v8h*)&Vt[vtbase + (size_t)row * Tdim + kBase + sub * 8];
    }
    __syncthreads();  // prior chunk's compute done before overwrite
#pragma unroll
    for (int i = 0; i < 4; ++i) {
      const int c = tid + 128 * i;
      const int row = c >> 3, sub = c & 7;
      *(v8h*)&sK[0][row][sub * 8] = rk[i];
      *(v8h*)&sV[0][row][sub * 8] = rv[i];
    }
    __syncthreads();
#endif

    // S^T = K @ Q^T : per 16-key tile j, accumulate over d in 2 steps of 32.
    v8f st[4];
#pragma unroll
    for (int j = 0; j < 4; ++j) {
      const _Float16* krow = &sK[cur][16 * j + r16][0];
      v16h ak0 = ldfrag(krow + 8 * g, krow + 16 + 8 * g);       // d 0..31
      v8f z = {};
      z = wmma_f16(ak0, bq[0], z);
      v16h ak1 = ldfrag(krow + 32 + 8 * g, krow + 48 + 8 * g);  // d 32..63
      st[j] = wmma_f16(ak1, bq[1], z);
    }

    // Lane-private online softmax for query r16; lane holds keys 16j+8g+r,
    // lanes q and q+16 split the 64 keys -> one xor-16 combine each.
    float lmax = -1.0e30f;
#pragma unroll
    for (int j = 0; j < 4; ++j)
#pragma unroll
      for (int r = 0; r < 8; ++r) lmax = fmaxf(lmax, st[j][r]);
    lmax = fmaxf(lmax, __shfl_xor(lmax, 16, 32));
    const float mnew = fmaxf(mI, lmax);
    const float alpha = __expf(mI - mnew);
    mI = mnew;

    float lsum = 0.0f;
#pragma unroll
    for (int j = 0; j < 4; ++j)
#pragma unroll
      for (int r = 0; r < 8; ++r) {
        const float e = __expf(st[j][r] - mnew);
        st[j][r] = e;
        lsum += e;
      }
    lsum += __shfl_xor(lsum, 16, 32);
    lI = lI * alpha + lsum;

#pragma unroll
    for (int n = 0; n < 4; ++n)
#pragma unroll
      for (int r = 0; r < 8; ++r) ot[n][r] *= alpha;

    // P^T (C/D layout) -> LDS as P[query][key]; lane's 8 values per j are
    // contiguous halves -> one packed ds_store_b128 per j.
#pragma unroll
    for (int j = 0; j < 4; ++j) {
      v8h ph;
#pragma unroll
      for (int r = 0; r < 8; ++r) ph[r] = (_Float16)st[j][r];
      *(v8h*)&sP[w][r16][16 * j + 8 * g] = ph;
    }

    // O^T += V^T @ P^T  (keys in 2 steps of 32)
#pragma unroll
    for (int j2 = 0; j2 < 2; ++j2) {
      const _Float16* prow = &sP[w][r16][32 * j2 + 8 * g];
      v16h bpf = ldfrag(prow, prow + 16);  // B-frag of P^T (col=query r16)
#pragma unroll
      for (int n = 0; n < 4; ++n) {
        const _Float16* vrow = &sV[cur][16 * n + r16][32 * j2];
        v16h av = ldfrag(vrow + 8 * g, vrow + 16 + 8 * g);  // A-frag of V^T
        ot[n] = wmma_f16(av, bpf, ot[n]);
      }
    }
  }

  // Normalize with lane-private 1/l and store: element (n, r) is
  // out[query qBase+r16][d = 16n + 8g + r] -> contiguous over r, packed b128.
  const float inv = 1.0f / lI;
  const size_t obase = (size_t)(bi * Tdim + qBase + r16) * Cdim + h * Dh;
#pragma unroll
  for (int n = 0; n < 4; ++n) {
    v8h oh;
#pragma unroll
    for (int r = 0; r < 8; ++r) oh[r] = (_Float16)(ot[n][r] * inv);
    *(v8h*)&O16[obase + 16 * n + 8 * g] = oh;
  }
}

// ---------------------------------------------------------------------------
// Host orchestration
// ---------------------------------------------------------------------------
extern "C" void kernel_launch(void* const* d_in, const int* in_sizes, int n_in,
                              void* d_out, int out_size, void* d_ws,
                              size_t ws_size, hipStream_t stream) {
  (void)in_sizes; (void)n_in; (void)out_size; (void)ws_size;

  const float* x  = (const float*)d_in[0];
  const float* Wk = (const float*)d_in[1];
  const float* bk = (const float*)d_in[2];
  const float* Wq = (const float*)d_in[3];
  const float* bq = (const float*)d_in[4];
  const float* Wv = (const float*)d_in[5];
  const float* bv = (const float*)d_in[6];
  const float* Wp = (const float*)d_in[7];
  const float* bp = (const float*)d_in[8];

  const int Mrows = 2 * Tdim;                 // B*T = 4096
  const size_t nx = (size_t)Mrows * Cdim;     // 4M halves
  const size_t nw = (size_t)Cdim * Cdim;      // 1M halves

  _Float16* p = (_Float16*)d_ws;
  _Float16* x16  = p; p += nx;   // reused for attention output
  _Float16* Wq16 = p; p += nw;
  _Float16* Wk16 = p; p += nw;
  _Float16* Wv16 = p; p += nw;
  _Float16* Wp16 = p; p += nw;
  _Float16* Q16  = p; p += nx;
  _Float16* K16  = p; p += nx;
  _Float16* Vt16 = p; p += nx;
  _Float16* A16  = x16;  // x16 is dead after the three projection GEMMs

  const float qscale = 0.125f;  // 1/sqrt(Dh), exact power of two

  k_convert_f16<<<(int)(nx / 256), 256, 0, stream>>>(x, x16, 1.0f, (int)nx);
  k_convert_f16<<<(int)(nw / 256), 256, 0, stream>>>(Wq, Wq16, qscale, (int)nw);
  k_convert_f16<<<(int)(nw / 256), 256, 0, stream>>>(Wk, Wk16, 1.0f, (int)nw);
  k_convert_f16<<<(int)(nw / 256), 256, 0, stream>>>(Wv, Wv16, 1.0f, (int)nw);
  k_convert_f16<<<(int)(nw / 256), 256, 0, stream>>>(Wp, Wp16, 1.0f, (int)nw);

  dim3 ggrid(Mrows / 64, Cdim / 64);
  k_gemm<0><<<ggrid, 128, 0, stream>>>(x16, Wq16, bq, qscale, (void*)Q16);
  k_gemm<0><<<ggrid, 128, 0, stream>>>(x16, Wk16, bk, 1.0f, (void*)K16);
  k_gemm<1><<<ggrid, 128, 0, stream>>>(x16, Wv16, bv, 1.0f, (void*)Vt16);

  dim3 agrid(Tdim / 64, Hn, 2);
  k_attention<<<agrid, 128, 0, stream>>>(Q16, K16, Vt16, A16);

  k_gemm<2><<<ggrid, 128, 0, stream>>>(A16, Wp16, bp, 1.0f, d_out);
}
